// MeanAggregator_83880711290996
// MI455X (gfx1250) — compile-verified, matrix operand order checked by reading, not run on hardware
//
#include <hip/hip_runtime.h>
#include <hip/hip_bf16.h>
#include <math.h>

// Problem constants (match reference)
#define D 256   // feature dim (K of GEMM)
#define A 128   // att size    (N of GEMM)
#define S 32    // neighbors   (M of GEMM, per node)
#define KT 8    // D / 32  (K steps per WMMA chain)
#define NT 8    // A / 16  (N tiles, one per wave)

typedef __attribute__((ext_vector_type(16))) __bf16 v16bf;
typedef __attribute__((ext_vector_type(8)))  float  v8f;

union BF16Frag { v16bf v; uint4 q[2]; };

__device__ __forceinline__ unsigned short f2bf(float f) {
  union { float f; unsigned u; } c; c.f = f;
  unsigned u = c.u;
  u += 0x7FFFu + ((u >> 16) & 1u);   // round-to-nearest-even to bf16
  return (unsigned short)(u >> 16);
}

// ---------------------------------------------------------------------------
// Prep: swizzle W_att (fp32 [D][A], row-major) into bf16 WMMA-B fragment
// layout: Wswz[kt][nt][lane][16 elems], so each lane's 16-element B fragment
// is one contiguous 32-byte run (two b128 loads).
// B-matrix (32x16, K x N) per-lane packing for 16x16x32 bf16:
//   lane 0-15 : N = lane,    elems 0..7 -> K = k0+0..7,  elems 8..15 -> K = k0+16..23
//   lane 16-31: N = lane-16, elems 0..7 -> K = k0+8..15, elems 8..15 -> K = k0+24..31
// ---------------------------------------------------------------------------
__global__ void han_prep_w(const float* __restrict__ W, unsigned short* __restrict__ Wswz) {
  int t = blockIdx.x * blockDim.x + threadIdx.x;
  if (t >= KT * NT * 32 * 16) return;
  int i    = t & 15;
  int lane = (t >> 4) & 31;
  int nt   = (t >> 9) & 7;
  int kt   = (t >> 12) & 7;
  int half = lane >> 4;
  int n    = nt * 16 + (lane & 15);
  int k    = (i < 8) ? (kt * 32 + half * 8 + i)
                     : (kt * 32 + 16 + half * 8 + (i - 8));
  Wswz[t] = f2bf(W[k * A + n]);
}

// ---------------------------------------------------------------------------
// Main: one block (8 waves, wave32) per node.
// ---------------------------------------------------------------------------
__global__ void __launch_bounds__(256)
han_attention(const float* __restrict__ features,
              const int*   __restrict__ neigh_idx,
              const unsigned short* __restrict__ Wswz,
              const float* __restrict__ b_att,
              const float* __restrict__ ctx_vec,
              float* __restrict__ out) {
  __shared__ __align__(16) unsigned short Xbf[S * D]; // 16 KB bf16 neighbor tile
  __shared__ float att[S];
  __shared__ float wgt[S];
  __shared__ int   nidx[S];

  const int b = blockIdx.x;
  const int t = threadIdx.x;

  // ---- Phase 0: indices + zero attention accumulators -----------------
  if (t < S) {
    nidx[t] = neigh_idx[b * S + t];
    att[t]  = 0.0f;
  }
  __syncthreads();

  // ---- Phase 1: gather + fp32->bf16 convert into LDS ------------------
  // thread t owns column d=t of every row: coalesced across the block.
  #pragma unroll 4
  for (int s = 0; s < S; ++s) {
    Xbf[s * D + t] = f2bf(features[(size_t)nidx[s] * D + t]);
  }
  __syncthreads();

  // ---- Phase 2: WMMA GEMM  h = X(32x256) * W(256x128), tanh, ctx dot --
  const int nt   = t >> 5;        // wave id == N tile
  const int lane = t & 31;
  const int half = lane >> 4;
  const int nl   = lane & 15;

  v8f acc0 = {};  // M tile 0 (rows 0..15)
  v8f acc1 = {};  // M tile 1 (rows 16..31)

  #pragma unroll
  for (int kt = 0; kt < KT; ++kt) {
    // B fragment: contiguous 32B per lane from pre-swizzled W
    BF16Frag bf;
    const uint4* bp = (const uint4*)(Wswz + (((kt * NT + nt) * 32 + lane) << 4));
    bf.q[0] = bp[0];
    bf.q[1] = bp[1];

    // A fragments from LDS: lane holds row M=nl; two 16B runs of K
    const int c0 = kt * 32 + half * 8;        // K run 0 start
    const int c1 = kt * 32 + 16 + half * 8;   // K run 1 start

    BF16Frag a0, a1;
    a0.q[0] = *(const uint4*)&Xbf[(0 * 16 + nl) * D + c0];
    a0.q[1] = *(const uint4*)&Xbf[(0 * 16 + nl) * D + c1];
    a1.q[0] = *(const uint4*)&Xbf[(1 * 16 + nl) * D + c0];
    a1.q[1] = *(const uint4*)&Xbf[(1 * 16 + nl) * D + c1];

    acc0 = __builtin_amdgcn_wmma_f32_16x16x32_bf16(false, a0.v, false, bf.v,
                                                   (short)0, acc0, false, false);
    acc1 = __builtin_amdgcn_wmma_f32_16x16x32_bf16(false, a1.v, false, bf.v,
                                                   (short)0, acc1, false, false);
  }

  // ---- Phase 3: bias + tanh + context dot + reduce to att[s] ----------
  // C/D layout: VGPR r, lanes 0-15 -> (M=r, N=lane); lanes 16-31 -> (M=r+8, N=lane-16)
  {
    const int  N  = nt * 16 + nl;
    const float bN = b_att[N];
    const float cN = ctx_vec[N];
    #pragma unroll
    for (int r = 0; r < 8; ++r) {
      float p0 = tanhf(acc0[r] + bN) * cN;
      float p1 = tanhf(acc1[r] + bN) * cN;
      // reduce over the 16 lanes of this half (covers N = nt*16 .. nt*16+15)
      #pragma unroll
      for (int off = 8; off > 0; off >>= 1) {
        p0 += __shfl_xor(p0, off, 16);
        p1 += __shfl_xor(p1, off, 16);
      }
      if (nl == 0) {
        const int M = r + half * 8;
        atomicAdd(&att[M],      p0);   // rows  0..15
        atomicAdd(&att[16 + M], p1);   // rows 16..31
      }
    }
  }
  __syncthreads();

  // ---- Phase 4: softmax over the 32 neighbors (one wave) --------------
  if (t < S) {
    float a = att[t];
    float m = a;
    #pragma unroll
    for (int off = 16; off > 0; off >>= 1) m = fmaxf(m, __shfl_xor(m, off, 32));
    float e = __expf(a - m);
    float sm = e;
    #pragma unroll
    for (int off = 16; off > 0; off >>= 1) sm += __shfl_xor(sm, off, 32);
    wgt[t] = e / sm;
  }
  __syncthreads();

  // ---- Phase 5: fp32 weighted sum (re-gather from L2-hot features) ----
  float acc = 0.0f;
  #pragma unroll 4
  for (int s = 0; s < S; ++s) {
    acc = fmaf(wgt[s], features[(size_t)nidx[s] * D + t], acc);
  }
  float o = acc * (1.0f / S);
  out[(size_t)b * D + t] = o > 0.0f ? o : 0.0f;
}

// ---------------------------------------------------------------------------
extern "C" void kernel_launch(void* const* d_in, const int* in_sizes, int n_in,
                              void* d_out, int out_size, void* d_ws, size_t ws_size,
                              hipStream_t stream) {
  const float* features  = (const float*)d_in[0];
  const int*   neigh_idx = (const int*)d_in[1];
  const float* W_att     = (const float*)d_in[2];
  const float* b_att     = (const float*)d_in[3];
  const float* ctx_vec   = (const float*)d_in[4];
  float*       out       = (float*)d_out;

  unsigned short* Wswz = (unsigned short*)d_ws;   // 64 KB bf16 swizzled W

  const int B = in_sizes[1] / S;                  // 20000 nodes

  const int prep_elems = KT * NT * 32 * 16;       // 32768
  han_prep_w<<<(prep_elems + 255) / 256, 256, 0, stream>>>(W_att, Wswz);
  han_attention<<<B, 256, 0, stream>>>(features, neigh_idx, Wswz,
                                       b_att, ctx_vec, out);
}
